// ConformerLayer_2516850835588
// MI455X (gfx1250) — matrix operand order repaired and checked by reading, not compiled, not run on hardware
//
#include <hip/hip_runtime.h>
#include <hip/hip_bf16.h>

// ---------------------------------------------------------------------------
// Conformer layer on gfx1250 (MI455X).  bf16 WMMA GEMMs (f32 accumulate)
// with double-buffered LDS fed by GLOBAL_LOAD_ASYNC_TO_LDS_B128 (ASYNCcnt),
// f32 residual stream, bandwidth-bound elementwise kernels.
// ---------------------------------------------------------------------------

typedef __attribute__((ext_vector_type(16))) __bf16 v16bf;
typedef __attribute__((ext_vector_type(8)))  float  v8f;

#define BTOT   (16 * 2048)        // B*T = 32768 rows
#define NB     16                 // batch
#define NT     2048               // time
#define ND     512                // model dim
#define NDFF   2048               // ff dim
#define NK     31                 // conv kernel
#define NPAD   15

// ---------------- workspace layout (bytes) ----------------
#define MB (1024ull * 1024ull)
static const size_t OFF_RESID = 0;                 // BT*D f32      = 64 MB
static const size_t OFF_HBF   = 64 * MB;           // BT*D bf16     = 32 MB
static const size_t OFF_T1    = 96 * MB;           // BT*DFF bf16   = 128 MB (also qkv 96MB, GLU-in 64MB)
static const size_t OFF_P     = 224 * MB;          // BT*D bf16     = 32 MB
static const size_t OFF_CONV  = 256 * MB;          // BT*D f32      = 64 MB
static const size_t OFF_KV    = 320 * MB;          // B*D f32       = 32 KB
static const size_t OFF_WB    = 321 * MB;          // bf16 weights ~= 11.5 MB

__device__ __forceinline__ float sigf(float x) {
    return 1.0f / (1.0f + __expf(-x));
}

// ---------------------------------------------------------------------------
// WMMA GEMM:  out[m,n] = epilogue( sum_k A[m,k] * W[n,k] + bias[n] )
//   A: (M x K) bf16 row-major, W: (N x K) bf16 row-major.
//   Block tile 128(M) x 128(N), 8 waves as 4(M) x 2(N), wave tile 32x64
//   (2x4 WMMA 16x16x32 bf16).  K-step 32, double-buffered LDS filled with
//   global_load_async_to_lds_b128 (ASYNCcnt-tracked, no VGPR staging).
//   Steady-state loop is branch-free: issue tile kt+1, wait asynccnt<=4
//   (drain tile kt), barrier, 12x ds_load_b128 + 8x WMMA, barrier.
//   MODE 0: store bf16 (val)       MODE 1: store bf16 swish(val)
//   MODE 3: out_f32 = res + scale*val
// ---------------------------------------------------------------------------
#define LDS_STRIDE 40              // 32 + 8 halfwords: 16B-aligned rows, spread banks
#define AELEM      (128 * LDS_STRIDE)        // elements per A (or B) tile buffer
#define BUFELEM    (2 * AELEM)               // A + B per buffer

struct Frag { union { uint4 u[2]; v16bf v; }; };

template <int MODE>
__global__ __launch_bounds__(256)
void gemm_wmma(const __bf16* __restrict__ A, const __bf16* __restrict__ W,
               const float* __restrict__ bias, void* __restrict__ outv,
               const float* __restrict__ res, float scale,
               int M, int N, int K)
{
    __shared__ __bf16 Smem[2 * BUFELEM];     // [buf][A 128x40 | B 128x40] = 40 KB

    const int tid  = threadIdx.x;
    const int lane = tid & 31;
    const int wv   = tid >> 5;
    const int wm   = wv & 3;                 // 4 waves along M (32 rows each)
    const int wn   = wv >> 2;                // 2 waves along N (64 cols each)
    const int bm   = blockIdx.y * 128;
    const int bn   = blockIdx.x * 128;

    const int m16  = lane & 15;
    const int hh   = lane >> 4;

    // per-thread async-copy assignment: 128 rows x 4 x 16B segments, 2 rows/thread
    const int arow = tid >> 2;               // 0..63 (and +64)
    const int aseg = tid & 3;

    const uint32_t lds_base = (uint32_t)(uintptr_t)&Smem[0];
    const uint32_t lseg     = (uint32_t)(arow * LDS_STRIDE + aseg * 8) * 2;  // bytes

    v8f acc[2][4] = {};

    const int ksteps = K >> 5;               // always >= 16 here

    // issue 4 async b128 loads for K-tile kt into LDS buffer `buf`
    auto issue_tile = [&](int kt, int buf) {
        const int k0 = kt << 5;
        const __bf16* ga0 = A + (size_t)(bm + arow)      * K + k0 + aseg * 8;
        const __bf16* ga1 = A + (size_t)(bm + arow + 64) * K + k0 + aseg * 8;
        const __bf16* gb0 = W + (size_t)(bn + arow)      * K + k0 + aseg * 8;
        const __bf16* gb1 = W + (size_t)(bn + arow + 64) * K + k0 + aseg * 8;
        const uint32_t la0 = lds_base + (uint32_t)(buf * BUFELEM) * 2 + lseg;
        const uint32_t la1 = la0 + (uint32_t)(64 * LDS_STRIDE) * 2;
        const uint32_t lb0 = la0 + (uint32_t)AELEM * 2;
        const uint32_t lb1 = lb0 + (uint32_t)(64 * LDS_STRIDE) * 2;
        asm volatile("global_load_async_to_lds_b128 %0, %1, off"
                     :: "v"(la0), "v"(ga0) : "memory");
        asm volatile("global_load_async_to_lds_b128 %0, %1, off"
                     :: "v"(la1), "v"(ga1) : "memory");
        asm volatile("global_load_async_to_lds_b128 %0, %1, off"
                     :: "v"(lb0), "v"(gb0) : "memory");
        asm volatile("global_load_async_to_lds_b128 %0, %1, off"
                     :: "v"(lb1), "v"(gb1) : "memory");
    };

    // 12x ds_load_b128 (fragment gather) + 8x WMMA from LDS buffer `buf`
    auto compute_tile = [&](int buf) {
        const __bf16* Ab = &Smem[buf * BUFELEM];
        const __bf16* Bb = Ab + AELEM;

        // ---- fragments (ISA 16-bit A 16x32 / B 32x16 lane layouts)
        Frag af[2], bf2[4];
#pragma unroll
        for (int mi = 0; mi < 2; ++mi) {
            const __bf16* p = &Ab[(wm * 32 + mi * 16 + m16) * LDS_STRIDE + hh * 8];
            af[mi].u[0] = *(const uint4*)(p);
            af[mi].u[1] = *(const uint4*)(p + 16);
        }
#pragma unroll
        for (int ni = 0; ni < 4; ++ni) {
            const __bf16* p = &Bb[(wn * 64 + ni * 16 + m16) * LDS_STRIDE + hh * 8];
            bf2[ni].u[0] = *(const uint4*)(p);
            bf2[ni].u[1] = *(const uint4*)(p + 16);
        }

#pragma unroll
        for (int mi = 0; mi < 2; ++mi)
#pragma unroll
            for (int ni = 0; ni < 4; ++ni)
                acc[mi][ni] = __builtin_amdgcn_wmma_f32_16x16x32_bf16(
                    false, af[mi].v, false, bf2[ni].v,
                    (short)0, acc[mi][ni], false, false);
    };

    issue_tile(0, 0);

    // steady state: branch-free pipeline
    for (int kt = 0; kt < ksteps - 1; ++kt) {
        const int buf = kt & 1;
        issue_tile(kt + 1, buf ^ 1);
        // drain the 4 older asyncs (tile kt); leave tile kt+1 in flight
        asm volatile("s_wait_asynccnt 0x4" ::: "memory");
        __syncthreads();                     // tile kt visible to all waves
        compute_tile(buf);
        __syncthreads();                     // reads done before buf is re-filled
    }
    // peeled last tile: full drain, no further issue
    asm volatile("s_wait_asynccnt 0x0" ::: "memory");
    __syncthreads();
    compute_tile((ksteps - 1) & 1);

    // ---- epilogue
#pragma unroll
    for (int mi = 0; mi < 2; ++mi) {
#pragma unroll
        for (int ni = 0; ni < 4; ++ni) {
            const int col = bn + wn * 64 + ni * 16 + m16;
            const float bc = bias[col];
#pragma unroll
            for (int r = 0; r < 8; ++r) {
                const int row = bm + wm * 32 + mi * 16 + hh * 8 + r;
                float v = acc[mi][ni][r] + bc;
                const size_t idx = (size_t)row * N + col;
                if (MODE == 0) {
                    ((__bf16*)outv)[idx] = (__bf16)v;
                } else if (MODE == 1) {
                    v = v * sigf(v);
                    ((__bf16*)outv)[idx] = (__bf16)v;
                } else { // MODE 3
                    ((float*)outv)[idx] = res[idx] + scale * v;
                }
            }
        }
    }
}

// ---------------------------------------------------------------------------
// LayerNorm over last dim (512), one 128-thread block per row.
// ---------------------------------------------------------------------------
template <bool OUTBF16, bool SWISH>
__global__ __launch_bounds__(128)
void ln_kernel(const float* __restrict__ in, const float* __restrict__ g,
               const float* __restrict__ b, void* __restrict__ outv)
{
    const int tid = threadIdx.x;
    const size_t base = (size_t)blockIdx.x * ND;
    float v[4], s = 0.f, ss = 0.f;
#pragma unroll
    for (int j = 0; j < 4; ++j) {
        v[j] = in[base + tid + j * 128];
        s += v[j]; ss += v[j] * v[j];
    }
    __shared__ float r1[128], r2[128];
    r1[tid] = s; r2[tid] = ss; __syncthreads();
    for (int off = 64; off > 0; off >>= 1) {
        if (tid < off) { r1[tid] += r1[tid + off]; r2[tid] += r2[tid + off]; }
        __syncthreads();
    }
    const float mean = r1[0] * (1.0f / ND);
    const float var  = r2[0] * (1.0f / ND) - mean * mean;
    const float rstd = rsqrtf(var + 1e-5f);
#pragma unroll
    for (int j = 0; j < 4; ++j) {
        const int idx = tid + j * 128;
        float y = (v[j] - mean) * rstd * g[idx] + b[idx];
        if (SWISH) y = y * sigf(y);
        if (OUTBF16) ((__bf16*)outv)[base + idx] = (__bf16)y;
        else         ((float*)outv)[base + idx]  = y;
    }
}

// ---------------------------------------------------------------------------
// Hydra attention helpers
// ---------------------------------------------------------------------------
__global__ __launch_bounds__(128)
void qknorm_kernel(__bf16* __restrict__ qkv)   // in-place L2 normalize q,k rows
{
    const int tid = threadIdx.x;
    const size_t base = (size_t)blockIdx.x * (3 * ND);
    float qv[4], kv[4], sq = 0.f, sk = 0.f;
#pragma unroll
    for (int j = 0; j < 4; ++j) {
        const int idx = tid + j * 128;
        qv[j] = (float)qkv[base + idx];
        kv[j] = (float)qkv[base + ND + idx];
        sq += qv[j] * qv[j]; sk += kv[j] * kv[j];
    }
    __shared__ float r1[128], r2[128];
    r1[tid] = sq; r2[tid] = sk; __syncthreads();
    for (int off = 64; off > 0; off >>= 1) {
        if (tid < off) { r1[tid] += r1[tid + off]; r2[tid] += r2[tid + off]; }
        __syncthreads();
    }
    const float rq = 1.0f / sqrtf(r1[0]);
    const float rk = 1.0f / sqrtf(r2[0]);
#pragma unroll
    for (int j = 0; j < 4; ++j) {
        const int idx = tid + j * 128;
        qkv[base + idx]      = (__bf16)(qv[j] * rq);
        qkv[base + ND + idx] = (__bf16)(kv[j] * rk);
    }
}

__global__ __launch_bounds__(256)
void kv_zero_kernel(float* __restrict__ kv)
{
    kv[blockIdx.x * 256 + threadIdx.x] = 0.0f;
}

__global__ __launch_bounds__(256)
void kv_reduce_kernel(const __bf16* __restrict__ qkv, float* __restrict__ kv)
{
    // grid = B * (T/128); each block accumulates 128 time rows
    const int b   = blockIdx.x >> 4;
    const int tch = blockIdx.x & 15;
    const int d0  = threadIdx.x * 2;
    float a0 = 0.f, a1 = 0.f;
    for (int tt = 0; tt < 128; ++tt) {
        const size_t base = (size_t)(b * NT + tch * 128 + tt) * (3 * ND);
        a0 += (float)qkv[base + ND + d0]     * (float)qkv[base + 2 * ND + d0];
        a1 += (float)qkv[base + ND + d0 + 1] * (float)qkv[base + 2 * ND + d0 + 1];
    }
    atomicAdd(&kv[b * ND + d0],     a0);
    atomicAdd(&kv[b * ND + d0 + 1], a1);
}

__global__ __launch_bounds__(256)
void attin_kernel(const __bf16* __restrict__ qkv, const float* __restrict__ kv,
                  __bf16* __restrict__ out)
{
    const size_t i  = (size_t)blockIdx.x * 256 + threadIdx.x;
    const size_t bt = i / ND;
    const int    d  = (int)(i % ND);
    const int    b  = (int)(bt / NT);
    out[i] = (__bf16)((float)qkv[bt * (3 * ND) + d] * kv[b * ND + d]);
}

// ---------------------------------------------------------------------------
// Conv module helpers
// ---------------------------------------------------------------------------
__global__ __launch_bounds__(256)
void glu_kernel(const __bf16* __restrict__ c2, __bf16* __restrict__ p)
{
    const size_t i  = (size_t)blockIdx.x * 256 + threadIdx.x;
    const size_t bt = i / ND;
    const int    d  = (int)(i % ND);
    const float a = (float)c2[bt * (2 * ND) + d];
    const float g = (float)c2[bt * (2 * ND) + ND + d];
    p[i] = (__bf16)(a * sigf(g));
}

__global__ __launch_bounds__(256)
void dwconv_kernel(const __bf16* __restrict__ p, const float* __restrict__ w,
                   const float* __restrict__ bias, float* __restrict__ out)
{
    const size_t i = (size_t)blockIdx.x * 256 + threadIdx.x;
    const int d = (int)(i % ND);
    const int t = (int)((i / ND) % NT);
    const int b = (int)(i / ((size_t)ND * NT));
    float acc = bias[d];
#pragma unroll
    for (int j = 0; j < NK; ++j) {
        const int tt = t + j - NPAD;
        if ((unsigned)tt < (unsigned)NT)
            acc += (float)p[((size_t)(b * NT + tt)) * ND + d] * w[d * NK + j];
    }
    out[i] = acc;
}

// ---------------------------------------------------------------------------
// f32 -> bf16 weight conversion
// ---------------------------------------------------------------------------
__global__ __launch_bounds__(256)
void cvt_bf16_kernel(const float* __restrict__ src, __bf16* __restrict__ dst, int n)
{
    const int i = blockIdx.x * 256 + threadIdx.x;
    if (i < n) dst[i] = (__bf16)src[i];
}

// ---------------------------------------------------------------------------
// Host side
// ---------------------------------------------------------------------------
static void launch_gemm(int mode, const __bf16* A, const __bf16* W, const float* bias,
                        void* out, const float* res, float scale,
                        int M, int N, int K, hipStream_t s)
{
    dim3 g(N / 128, M / 128), b(256);
    if (mode == 0)      gemm_wmma<0><<<g, b, 0, s>>>(A, W, bias, out, res, scale, M, N, K);
    else if (mode == 1) gemm_wmma<1><<<g, b, 0, s>>>(A, W, bias, out, res, scale, M, N, K);
    else                gemm_wmma<3><<<g, b, 0, s>>>(A, W, bias, out, res, scale, M, N, K);
}

extern "C" void kernel_launch(void* const* d_in, const int* in_sizes, int n_in,
                              void* d_out, int out_size, void* d_ws, size_t ws_size,
                              hipStream_t stream)
{
    const float* x      = (const float*)d_in[0];
    const float* ln1_g  = (const float*)d_in[1];
    const float* ln1_b  = (const float*)d_in[2];
    const float* ff1_w1 = (const float*)d_in[3];
    const float* ff1_b1 = (const float*)d_in[4];
    const float* ff1_w2 = (const float*)d_in[5];
    const float* ff1_b2 = (const float*)d_in[6];
    const float* lna_g  = (const float*)d_in[7];
    const float* lna_b  = (const float*)d_in[8];
    const float* qkv_w  = (const float*)d_in[9];
    const float* qkv_b  = (const float*)d_in[10];
    const float* ao_w   = (const float*)d_in[11];
    const float* ao_b   = (const float*)d_in[12];
    const float* lnc_g  = (const float*)d_in[13];
    const float* lnc_b  = (const float*)d_in[14];
    const float* pw1_w  = (const float*)d_in[15];
    const float* pw1_b  = (const float*)d_in[16];
    const float* dw_w   = (const float*)d_in[17];
    const float* dw_b   = (const float*)d_in[18];
    const float* lncn_g = (const float*)d_in[19];
    const float* lncn_b = (const float*)d_in[20];
    const float* pw2_w  = (const float*)d_in[21];
    const float* pw2_b  = (const float*)d_in[22];
    const float* ln2_g  = (const float*)d_in[23];
    const float* ln2_b  = (const float*)d_in[24];
    const float* ff2_w1 = (const float*)d_in[25];
    const float* ff2_b1 = (const float*)d_in[26];
    const float* ff2_w2 = (const float*)d_in[27];
    const float* ff2_b2 = (const float*)d_in[28];
    const float* lno_g  = (const float*)d_in[29];
    const float* lno_b  = (const float*)d_in[30];

    char* ws = (char*)d_ws;
    float*  resid   = (float*)(ws + OFF_RESID);
    __bf16* hbf     = (__bf16*)(ws + OFF_HBF);
    __bf16* t1      = (__bf16*)(ws + OFF_T1);     // big bf16 scratch (t1 / qkv / glu-in)
    __bf16* pbuf    = (__bf16*)(ws + OFF_P);
    float*  convout = (float*)(ws + OFF_CONV);
    float*  kvbuf   = (float*)(ws + OFF_KV);
    __bf16* wb      = (__bf16*)(ws + OFF_WB);

    // bf16 weight slots
    __bf16* wb_ff1w1 = wb;                        // DFF*D
    __bf16* wb_ff1w2 = wb_ff1w1 + NDFF * ND;      // D*DFF
    __bf16* wb_qkv   = wb_ff1w2 + ND * NDFF;      // 3D*D
    __bf16* wb_ao    = wb_qkv   + 3 * ND * ND;    // D*D
    __bf16* wb_pw1   = wb_ao    + ND * ND;        // 2D*D
    __bf16* wb_pw2   = wb_pw1   + 2 * ND * ND;    // D*D
    __bf16* wb_ff2w1 = wb_pw2   + ND * ND;        // DFF*D
    __bf16* wb_ff2w2 = wb_ff2w1 + NDFF * ND;      // D*DFF

    auto cvt = [&](const float* s, __bf16* d, int n) {
        cvt_bf16_kernel<<<(n + 255) / 256, 256, 0, stream>>>(s, d, n);
    };
    cvt(ff1_w1, wb_ff1w1, NDFF * ND);
    cvt(ff1_w2, wb_ff1w2, ND * NDFF);
    cvt(qkv_w,  wb_qkv,   3 * ND * ND);
    cvt(ao_w,   wb_ao,    ND * ND);
    cvt(pw1_w,  wb_pw1,   2 * ND * ND);
    cvt(pw2_w,  wb_pw2,   ND * ND);
    cvt(ff2_w1, wb_ff2w1, NDFF * ND);
    cvt(ff2_w2, wb_ff2w2, ND * NDFF);

    const int EW = BTOT * ND / 256;   // elementwise grid over BT*D

    // ---- FF1 half-step: resid = x + 0.5 * FFN(LN(x))
    ln_kernel<true, false><<<BTOT, 128, 0, stream>>>(x, ln1_g, ln1_b, hbf);
    launch_gemm(1, hbf, wb_ff1w1, ff1_b1, t1, nullptr, 0.f, BTOT, NDFF, ND, stream);
    launch_gemm(3, t1, wb_ff1w2, ff1_b2, resid, x, 0.5f, BTOT, ND, NDFF, stream);

    // ---- Hydra attention: resid += Wo((q/|q|) * sum_t((k/|k|)*v)) + b
    ln_kernel<true, false><<<BTOT, 128, 0, stream>>>(resid, lna_g, lna_b, hbf);
    launch_gemm(0, hbf, wb_qkv, qkv_b, t1, nullptr, 0.f, BTOT, 3 * ND, ND, stream);
    qknorm_kernel<<<BTOT, 128, 0, stream>>>(t1);
    kv_zero_kernel<<<NB * ND / 256, 256, 0, stream>>>(kvbuf);
    kv_reduce_kernel<<<NB * (NT / 128), 256, 0, stream>>>(t1, kvbuf);
    attin_kernel<<<EW, 256, 0, stream>>>(t1, kvbuf, hbf);
    launch_gemm(3, hbf, wb_ao, ao_b, resid, resid, 1.0f, BTOT, ND, ND, stream);

    // ---- Conv module
    ln_kernel<true, false><<<BTOT, 128, 0, stream>>>(resid, lnc_g, lnc_b, hbf);
    launch_gemm(0, hbf, wb_pw1, pw1_b, t1, nullptr, 0.f, BTOT, 2 * ND, ND, stream);
    glu_kernel<<<EW, 256, 0, stream>>>(t1, pbuf);
    dwconv_kernel<<<EW, 256, 0, stream>>>(pbuf, dw_w, dw_b, convout);
    ln_kernel<true, true><<<BTOT, 128, 0, stream>>>(convout, lncn_g, lncn_b, hbf);
    launch_gemm(3, hbf, wb_pw2, pw2_b, resid, resid, 1.0f, BTOT, ND, ND, stream);

    // ---- FF2 half-step
    ln_kernel<true, false><<<BTOT, 128, 0, stream>>>(resid, ln2_g, ln2_b, hbf);
    launch_gemm(1, hbf, wb_ff2w1, ff2_b1, t1, nullptr, 0.f, BTOT, NDFF, ND, stream);
    launch_gemm(3, t1, wb_ff2w2, ff2_b2, resid, resid, 0.5f, BTOT, ND, NDFF, stream);

    // ---- final LN -> f32 output
    ln_kernel<false, false><<<BTOT, 128, 0, stream>>>(resid, lno_g, lno_b, (float*)d_out);

    (void)in_sizes; (void)n_in; (void)out_size; (void)ws_size;
}
